// RFHAR_74053826117642
// MI455X (gfx1250) — matrix operand order repaired and checked by reading, not compiled, not run on hardware
//
#include <hip/hip_runtime.h>
#include <hip/hip_bf16.h>
#include <stdint.h>

// ---------------- problem constants (match reference) ----------------
#define B_SZ     16
#define N_HEADS  32
#define K_FULL   32768
#define K_IMG    16384
#define GAMMA    0.3f
#define LPEN     0.5f
#define EPSV     1e-6f
#define K_HEADS  7          // ceil(0.2 * 32)
#define KSLICES  32
#define KCHUNK   (K_IMG / KSLICES)   // 512 image tokens per wave-slice
#define ASCALE   1024.0f

typedef __attribute__((ext_vector_type(16))) _Float16 v16h;
typedef __attribute__((ext_vector_type(8)))  _Float16 v8h;
typedef __attribute__((ext_vector_type(8)))  float    v8f;
typedef __attribute__((ext_vector_type(4)))  uint32_t v4u;
typedef __attribute__((ext_vector_type(8)))  uint32_t v8u;

__device__ __forceinline__ float sigmoidf_(float z) {
    return 1.0f / (1.0f + __expf(-z));
}

// =====================================================================
// Kernel 1: rf[b,k] (f32) plus f16 copies of rf and max(1-rf,0)
// =====================================================================
__global__ __launch_bounds__(256)
void k_rf(const float* __restrict__ fC, const float* __restrict__ fA,
          const float* __restrict__ fD, const float* __restrict__ fB,
          float* __restrict__ rf, _Float16* __restrict__ rf16,
          _Float16* __restrict__ low16) {
    const int b   = blockIdx.x;
    const int tid = threadIdx.x;
    __shared__ float red[256];
    __shared__ float smu[4], sistd[4];

    const float* xs[4] = { fC + (size_t)b * K_IMG, fA + (size_t)b * K_IMG,
                           fD + (size_t)b * K_IMG, fB + (size_t)b * K_IMG };

    for (int f = 0; f < 4; ++f) {
        const float* x = xs[f];
        float s = 0.f, s2 = 0.f;
        for (int k = tid; k < K_IMG; k += 256) {
            float v = x[k];
            s += v; s2 += v * v;
        }
        red[tid] = s; __syncthreads();
        for (int off = 128; off > 0; off >>= 1) {
            if (tid < off) red[tid] += red[tid + off];
            __syncthreads();
        }
        if (tid == 0) smu[f] = red[0] * (1.0f / K_IMG);
        __syncthreads();
        red[tid] = s2; __syncthreads();
        for (int off = 128; off > 0; off >>= 1) {
            if (tid < off) red[tid] += red[tid + off];
            __syncthreads();
        }
        if (tid == 0) {
            float mu  = smu[f];
            float var = red[0] * (1.0f / K_IMG) - mu * mu;   // biased
            sistd[f]  = 1.0f / (sqrtf(fmaxf(var, 0.f)) + EPSV);
        }
        __syncthreads();
    }

    for (int k = tid; k < K_IMG; k += 256) {
        float Ct = fmaxf((xs[0][k] - smu[0]) * sistd[0], 0.f);
        float At = sigmoidf_((xs[1][k] - smu[1]) * sistd[1]);
        float Dt = sigmoidf_((xs[2][k] - smu[2]) * sistd[2]);
        float Bt = sigmoidf_((xs[3][k] - smu[3]) * sistd[3]);
        float denom = fmaxf(1.0f + LPEN * (Dt + Bt), EPSV);
        float v = fmaxf(Ct * At / denom, 0.f);
        rf   [(size_t)b * K_IMG + k] = v;
        rf16 [(size_t)b * K_IMG + k] = (_Float16)v;
        low16[(size_t)b * K_IMG + k] = (_Float16)fmaxf(1.0f - v, 0.f);
    }
}

// =====================================================================
// Kernel 2: prefix-scan of image_mask -> idx[b,j] and rank[b,p]
// =====================================================================
__global__ __launch_bounds__(256)
void k_scan(const int* __restrict__ mask, int* __restrict__ idxArr,
            int* __restrict__ rankArr) {
    const int b   = blockIdx.x;
    const int tid = threadIdx.x;
    __shared__ int sd[256];
    const int* mrow = mask + (size_t)b * K_FULL;
    int carry = 0;
    for (int base = 0; base < K_FULL; base += 256) {
        int p = base + tid;
        int v = mrow[p];
        sd[tid] = v; __syncthreads();
        for (int off = 1; off < 256; off <<= 1) {
            int add = (tid >= off) ? sd[tid - off] : 0;
            __syncthreads();
            sd[tid] += add;
            __syncthreads();
        }
        int excl = sd[tid] - v;
        if (v) {
            int r = carry + excl;
            idxArr[(size_t)b * K_IMG + r]   = p;
            rankArr[(size_t)b * K_FULL + p] = r;
        } else {
            rankArr[(size_t)b * K_FULL + p] = -1;
        }
        carry += sd[255];
        __syncthreads();
    }
}

// =====================================================================
// Kernel 3: per-(b,h) softmax stats -> rowmax, scale
// =====================================================================
__global__ __launch_bounds__(256)
void k_stats(const float* __restrict__ logits, const int* __restrict__ mask,
             float* __restrict__ rowmax, float* __restrict__ scale) {
    const int bh  = blockIdx.x;
    const int b   = bh / N_HEADS;
    const int tid = threadIdx.x;
    const float* row  = logits + (size_t)bh * K_FULL;
    const int*   mrow = mask   + (size_t)b  * K_FULL;
    __shared__ float red[256];

    float lmax = -__builtin_inff();
    for (int p = tid; p < K_FULL; p += 256) {
        __builtin_prefetch(row + p + 4096, 0, 0);   // global_prefetch_b8
        lmax = fmaxf(lmax, row[p]);
    }
    red[tid] = lmax; __syncthreads();
    for (int off = 128; off > 0; off >>= 1) {
        if (tid < off) red[tid] = fmaxf(red[tid], red[tid + off]);
        __syncthreads();
    }
    float rmax = red[0];
    __syncthreads();

    float se = 0.f, si = 0.f;
    for (int p = tid; p < K_FULL; p += 256) {
        float e = __expf(row[p] - rmax);
        se += e;
        si += mrow[p] ? e : 0.f;
    }
    red[tid] = se; __syncthreads();
    for (int off = 128; off > 0; off >>= 1) {
        if (tid < off) red[tid] += red[tid + off];
        __syncthreads();
    }
    float denom_full = red[0];
    __syncthreads();
    red[tid] = si; __syncthreads();
    for (int off = 128; off > 0; off >>= 1) {
        if (tid < off) red[tid] += red[tid + off];
        __syncthreads();
    }
    if (tid == 0) {
        float denom_img = red[0];
        rowmax[bh] = rmax;
        scale[bh]  = 1.0f / (denom_full * fmaxf(denom_img / denom_full, EPSV));
    }
}

// =====================================================================
// Kernel 4: gather + exp -> pimg[b,h,j] = f16(attn_img_norm * 2^10), contiguous
// =====================================================================
__global__ __launch_bounds__(256)
void k_gather(const float* __restrict__ logits, const int* __restrict__ idxArr,
              const float* __restrict__ rowmax, const float* __restrict__ scale,
              _Float16* __restrict__ pimg) {
    int gid  = blockIdx.x * 256 + threadIdx.x;      // B*H*K_IMG/8 threads
    int jblk = gid & (K_IMG / 8 - 1);
    int bh   = gid >> 11;                           // log2(K_IMG/8) = 11
    int b    = bh >> 5;
    int j0   = jblk * 8;
    const float rmax = rowmax[bh];
    const float sc   = scale[bh] * ASCALE;
    const float* lrow = logits + (size_t)bh * K_FULL;
    const int*   idxb = idxArr + (size_t)b * K_IMG;
    __builtin_prefetch(idxb + j0 + 512, 0, 0);
    v8h o;
#pragma unroll
    for (int t = 0; t < 8; ++t) {
        int p = idxb[j0 + t];
        o[t] = (_Float16)(__expf(lrow[p] - rmax) * sc);
    }
    *(v8h*)(pimg + (size_t)bh * K_IMG + j0) = o;
}

// =====================================================================
// Kernel 5: zero s_pos / s_neg
// =====================================================================
__global__ void k_zero(float* __restrict__ sp, float* __restrict__ sn) {
    int i = blockIdx.x * blockDim.x + threadIdx.x;
    if (i < B_SZ * N_HEADS) { sp[i] = 0.f; sn[i] = 0.f; }
}

// =====================================================================
// Kernel 6 (WMMA + TDM + async): S = P[16h x K] x R^T[K x 16]
//   s_B regions: [0]=rf16 (TDM tensor_load_to_lds), [1]=low16
//   (global_load_async_to_lds_b128), [2]=zeros (ds stores).
//   Loop is fully convergent: every lane reads its B fragment from
//   s_B + min(m,2)*KCHUNK, so no exec-mask juggling around the WMMAs.
// =====================================================================
__global__ __launch_bounds__(32)
void k_scores_wmma(const _Float16* __restrict__ pimg,
                   const _Float16* __restrict__ rf16,
                   const _Float16* __restrict__ low16,
                   float* __restrict__ s_pos, float* __restrict__ s_neg) {
    __shared__ alignas(64) _Float16 s_B[3 * KCHUNK];   // 3 KB

    const int blk   = blockIdx.x;
    const int slice = blk % KSLICES;
    const int tile  = (blk / KSLICES) & 1;
    const int b     = blk / (KSLICES * 2);
    const int lane  = threadIdx.x;
    const int m     = lane & 15;
    const int half  = lane >> 4;

    const int jbase = slice * KCHUNK;
    const _Float16* rfrow  = rf16  + (size_t)b * K_IMG + jbase;
    const _Float16* lowrow = low16 + (size_t)b * K_IMG + jbase;

    // flat LDS address truncated to 32 bits == LDS byte offset (ISA 10.2)
    uint32_t b_lds = (uint32_t)(uintptr_t)&s_B[0];

    // ---- TDM: rf16 chunk -> region 0 ---------------------------------
    {
        uint64_t ga = (uint64_t)(uintptr_t)rfrow;
        v4u g0; v8u g1;
        g0[0] = 1u;                                     // count=1, user descriptor
        g0[1] = b_lds;                                  // lds_addr (region 0)
        g0[2] = (uint32_t)ga;                           // global_addr[31:0]
        g0[3] = (uint32_t)((ga >> 32) & 0x1ffffffu) | (2u << 30);  // addr[56:32] | type=2
        g1[0] = (1u << 16);                             // data_size=1 (2 bytes)
        g1[1] = ((uint32_t)KCHUNK & 0xffffu) << 16;     // tensor_dim0 lo16
        g1[2] = ((uint32_t)KCHUNK >> 16) | (1u << 16);  // tensor_dim0 hi | tensor_dim1 lo
        g1[3] = ((uint32_t)KCHUNK << 16);               // tensor_dim1 hi | tile_dim0
        g1[4] = 1u;                                     // tile_dim1=1, tile_dim2=0
        g1[5] = (uint32_t)KCHUNK;                       // tensor_dim0_stride lo32
        g1[6] = 0u;                                     // stride0 hi | stride1 lo
        g1[7] = 0u;                                     // stride1 hi
        asm volatile("tensor_load_to_lds %0, %1" :: "s"(g0), "s"(g1) : "memory");
    }

    // ---- async: low16 chunk -> region 1 (2 x b128 per lane) ----------
    {
        uint32_t l0  = b_lds + (uint32_t)KCHUNK * 2u + (uint32_t)lane * 16u;
        uint64_t g0a = (uint64_t)(uintptr_t)(lowrow) + (uint64_t)lane * 16u;
        asm volatile("global_load_async_to_lds_b128 %0, %1, off"
                     :: "v"(l0), "v"(g0a) : "memory");
        uint32_t l1  = l0 + 512u;
        uint64_t g1a = g0a + 512u;
        asm volatile("global_load_async_to_lds_b128 %0, %1, off"
                     :: "v"(l1), "v"(g1a) : "memory");
    }

    // ---- region 2 = zeros (one v16h ds store per lane) ---------------
    {
        v16h z = {};
        *(v16h*)(s_B + 2 * KCHUNK + lane * 16) = z;
    }

    __builtin_amdgcn_s_wait_tensorcnt(0);
    asm volatile("s_wait_asynccnt 0x0" ::: "memory");
    __syncthreads();

    const _Float16* arow =
        pimg + ((size_t)b * N_HEADS + tile * 16 + m) * K_IMG + jbase;
    const _Float16* bsrc = s_B + (m < 2 ? m : 2) * KCHUNK;

    v8f c = {};
#pragma unroll 8
    for (int j0 = 0; j0 < KCHUNK; j0 += 32) {
        // A fragment: halves 0..15 -> K=half*16+t (contig 16 halves)
        v8h alo = *(const v8h*)(arow + j0 + half * 16);
        v8h ahi = *(const v8h*)(arow + j0 + half * 16 + 8);
        v16h a;
#pragma unroll
        for (int t = 0; t < 8; ++t) { a[t] = alo[t]; a[t + 8] = ahi[t]; }
        // B fragment: lanes 0-15 K=t, lanes 16-31 K=16+t (contig 16 halves)
        v16h bb = *(const v16h*)(bsrc + j0 + half * 16);
        c = __builtin_amdgcn_wmma_f32_16x16x32_f16(
                false, a, false, bb, (short)0, c, false, false);
    }

    // C/D f32 16x16: lane<16 -> N=lane, M=v ; lane>=16 -> N=lane-16, M=v+8
    if (m < 2) {
        float* dst = (m == 0) ? s_pos : s_neg;
#pragma unroll
        for (int v = 0; v < 8; ++v) {
            int M = v + half * 8;
            atomicAdd(&dst[b * N_HEADS + tile * 16 + M], c[v] * (1.0f / ASCALE));
        }
    }
}

// =====================================================================
// Kernel 7: top-k head roles (H=32, k=7), serial per batch
// =====================================================================
__global__ void k_roles(const float* __restrict__ s_pos, const float* __restrict__ s_neg,
                        float* __restrict__ m_pos, float* __restrict__ m_neg) {
    const int b = blockIdx.x;
    if (threadIdx.x != 0) return;
    float sp[N_HEADS], sn[N_HEADS];
    int chosen[N_HEADS], mp[N_HEADS], mn[N_HEADS];
    for (int h = 0; h < N_HEADS; ++h) {
        sp[h] = s_pos[b * N_HEADS + h];
        sn[h] = s_neg[b * N_HEADS + h];
        chosen[h] = 0; mp[h] = 0; mn[h] = 0;
    }
    for (int it = 0; it < K_HEADS; ++it) {
        float best = -__builtin_inff(); int bi = -1;
        for (int h = 0; h < N_HEADS; ++h)
            if (!chosen[h] && sp[h] > best) { best = sp[h]; bi = h; }
        if (bi < 0) break;
        chosen[bi] = 1;
        if (best > 0.f) mp[bi] = 1;
    }
    float ns[N_HEADS];
    for (int h = 0; h < N_HEADS; ++h) {
        ns[h] = (mp[h] > 0) ? -__builtin_inff() : sn[h];
        chosen[h] = 0;
    }
    for (int it = 0; it < K_HEADS; ++it) {
        float best = -__builtin_inff(); int bi = -1;
        for (int h = 0; h < N_HEADS; ++h)
            if (!chosen[h] && ns[h] > best) { best = ns[h]; bi = h; }
        if (bi < 0) break;
        chosen[bi] = 1;
        if (best > 0.f && mp[bi] == 0) mn[bi] = 1;
    }
    for (int h = 0; h < N_HEADS; ++h) {
        m_pos[b * N_HEADS + h] = (float)mp[h];
        m_neg[b * N_HEADS + h] = (float)mn[h];
    }
}

// =====================================================================
// Kernel 8: streaming apply, float4/int4 vectorized (b128 traffic)
// =====================================================================
__global__ __launch_bounds__(256)
void k_apply(const float* __restrict__ logits, const int* __restrict__ mask,
             const int* __restrict__ rankArr, const float* __restrict__ rf,
             const float* __restrict__ m_pos, const float* __restrict__ m_neg,
             float* __restrict__ out) {
    int t4 = blockIdx.x * 256 + threadIdx.x;
    int i  = t4 * 4;                                // total = 2^24 elements
    __builtin_prefetch(logits + i + 16384, 0, 0);   // global_prefetch_b8
    int p  = i & (K_FULL - 1);
    int bh = i >> 15;
    int b  = bh >> 5;
    float4 v = *(const float4*)(logits + i);
    int4  mk = *(const int4*)(mask + (size_t)b * K_FULL + p);
    float mp = m_pos[bh], mn = m_neg[bh];
    if ((mp != 0.f || mn != 0.f) && (mk.x | mk.y | mk.z | mk.w)) {
        int4 rk = *(const int4*)(rankArr + (size_t)b * K_FULL + p);
        const float* rfb = rf + (size_t)b * K_IMG;
        if (mk.x) { float rv = rfb[rk.x]; v.x += GAMMA * (mp * rv - mn * fmaxf(1.f - rv, 0.f)); }
        if (mk.y) { float rv = rfb[rk.y]; v.y += GAMMA * (mp * rv - mn * fmaxf(1.f - rv, 0.f)); }
        if (mk.z) { float rv = rfb[rk.z]; v.z += GAMMA * (mp * rv - mn * fmaxf(1.f - rv, 0.f)); }
        if (mk.w) { float rv = rfb[rk.w]; v.w += GAMMA * (mp * rv - mn * fmaxf(1.f - rv, 0.f)); }
    }
    *(float4*)(out + i) = v;
}

// =====================================================================
// launcher
// =====================================================================
extern "C" void kernel_launch(void* const* d_in, const int* in_sizes, int n_in,
                              void* d_out, int out_size, void* d_ws, size_t ws_size,
                              hipStream_t stream) {
    const float* logits = (const float*)d_in[0];   // [B,H,K_FULL]
    const int*   mask   = (const int*)  d_in[1];   // [B,K_FULL]
    const float* fC     = (const float*)d_in[2];
    const float* fA     = (const float*)d_in[3];
    const float* fD     = (const float*)d_in[4];
    const float* fB     = (const float*)d_in[5];
    float*       out    = (float*)d_out;

    // carve workspace (~21.3 MB)
    char* w = (char*)d_ws;
    auto carve = [&](size_t bytes) -> void* {
        void* p = (void*)w;
        w += (bytes + 255) & ~(size_t)255;
        return p;
    };
    float*     rf      = (float*)    carve((size_t)B_SZ * K_IMG  * sizeof(float));
    _Float16*  rf16    = (_Float16*) carve((size_t)B_SZ * K_IMG  * sizeof(_Float16));
    _Float16*  low16   = (_Float16*) carve((size_t)B_SZ * K_IMG  * sizeof(_Float16));
    _Float16*  pimg    = (_Float16*) carve((size_t)B_SZ * N_HEADS * K_IMG * sizeof(_Float16));
    int*       rankArr = (int*)      carve((size_t)B_SZ * K_FULL * sizeof(int));
    int*       idxArr  = (int*)      carve((size_t)B_SZ * K_IMG  * sizeof(int));
    float*     rowmax  = (float*)    carve((size_t)B_SZ * N_HEADS * sizeof(float));
    float*     scale   = (float*)    carve((size_t)B_SZ * N_HEADS * sizeof(float));
    float*     s_pos   = (float*)    carve((size_t)B_SZ * N_HEADS * sizeof(float));
    float*     s_neg   = (float*)    carve((size_t)B_SZ * N_HEADS * sizeof(float));
    float*     m_pos   = (float*)    carve((size_t)B_SZ * N_HEADS * sizeof(float));
    float*     m_neg   = (float*)    carve((size_t)B_SZ * N_HEADS * sizeof(float));

    k_rf    <<<B_SZ,           256, 0, stream>>>(fC, fA, fD, fB, rf, rf16, low16);
    k_scan  <<<B_SZ,           256, 0, stream>>>(mask, idxArr, rankArr);
    k_stats <<<B_SZ * N_HEADS, 256, 0, stream>>>(logits, mask, rowmax, scale);
    k_gather<<<(B_SZ * N_HEADS * (K_IMG / 8)) / 256, 256, 0, stream>>>(
        logits, idxArr, rowmax, scale, pimg);
    k_zero  <<<(B_SZ * N_HEADS + 255) / 256, 256, 0, stream>>>(s_pos, s_neg);
    k_scores_wmma<<<B_SZ * 2 * KSLICES, 32, 0, stream>>>(pimg, rf16, low16,
                                                         s_pos, s_neg);
    k_roles <<<B_SZ, 32, 0, stream>>>(s_pos, s_neg, m_pos, m_neg);
    k_apply <<<(B_SZ * N_HEADS * K_FULL / 4) / 256, 256, 0, stream>>>(
        logits, mask, rankArr, rf, m_pos, m_neg, out);
}
